// SelfAttentionHead_30133490549449
// MI455X (gfx1250) — matrix operand order repaired and checked by reading, not compiled, not run on hardware
//
#include <hip/hip_runtime.h>
#include <math.h>

typedef __attribute__((ext_vector_type(16))) __bf16 v16bf;
typedef __attribute__((ext_vector_type(8)))  __bf16 v8bf;
typedef __attribute__((ext_vector_type(8)))  float  v8f;
typedef __attribute__((ext_vector_type(4)))  float  v4f;

#if __has_builtin(__builtin_amdgcn_exp2f)
#define EXP2F(x) __builtin_amdgcn_exp2f(x)
#else
#define EXP2F(x) exp2f(x)
#endif

#if __has_builtin(__builtin_amdgcn_s_wait_asynccnt)
#define WAIT_ASYNC() __builtin_amdgcn_s_wait_asynccnt(0)
#else
#define WAIT_ASYNC() asm volatile("s_wait_asynccnt 0" ::: "memory")
#endif

// Async global->LDS (CDNA5, ASYNCcnt-tracked). vdst = LDS byte offset VGPR,
// vaddr = 64-bit global address VGPR pair (GV mode).
#define HAS_ASYNC_LDS 1

__device__ __forceinline__ void stage_granule16(const __bf16* g, unsigned lds_off,
                                                __bf16* lds_ptr) {
#if HAS_ASYNC_LDS
  asm volatile("global_load_async_to_lds_b128 %0, %1, off"
               :: "v"(lds_off), "v"(g) : "memory");
#else
  v8bf t = *(const v8bf*)g;
  *(v8bf*)lds_ptr = t;
#endif
}

static constexpr int BB = 8;
static constexpr int TT = 2048;
static constexpr int CC = 2048;
static constexpr int HH = 128;

// ---------------------------------------------------------------------------
// Kernel 0: transpose + convert weights: W[C,H] f32 -> Wt[H,C] bf16  (x3 mats)
// ---------------------------------------------------------------------------
__global__ __launch_bounds__(256) void wtrans_kernel(
    const float* __restrict__ Wq, const float* __restrict__ Wk,
    const float* __restrict__ Wv, __bf16* __restrict__ Wt) {
  int tid = blockIdx.x * 256 + threadIdx.x;      // 3*C*H threads total
  int mat = tid / (CC * HH);
  int rem = tid - mat * (CC * HH);
  int c = rem / HH;
  int h = rem - c * HH;
  const float* W = (mat == 0) ? Wq : ((mat == 1) ? Wk : Wv);
  Wt[(size_t)mat * CC * HH + (size_t)h * CC + c] = (__bf16)W[rem];
}

// ---------------------------------------------------------------------------
// Kernel 1: QKV projection. Block = 8 waves, one matrix, 128 rows of x.
// Each wave computes a 16x128 strip (8 accumulator tiles). The 32x128 W chunk
// for the current k-step is staged in LDS (double-buffered) via async
// global->LDS loads and shared by all 8 waves.
// Q,K stored [B*T, H] bf16 row-major; V stored transposed per batch: [H, T].
// ---------------------------------------------------------------------------
__global__ __launch_bounds__(256) void qkv_kernel(
    const float* __restrict__ x, const __bf16* __restrict__ Wt,
    const float* __restrict__ bq, const float* __restrict__ bk,
    const float* __restrict__ bv,
    __bf16* __restrict__ Q, __bf16* __restrict__ K, __bf16* __restrict__ Vt) {
  // Two 8KB buffers, each holding W chunk as [h(128)][k(32)] bf16.
  __shared__ __align__(16) __bf16 wbuf[2][128 * 32];

  int w    = threadIdx.x >> 5;
  int lane = threadIdx.x & 31;
  int l16  = lane & 15;
  int hi   = lane >> 4;

  int mat  = blockIdx.x >> 7;          // 128 blocks per matrix
  int grp  = blockIdx.x & 127;
  int row0 = grp * 128 + w * 16;       // this wave's 16 rows in [0, B*T)

  const __bf16* Wm   = Wt + (size_t)mat * CC * HH;   // [H][C] bf16
  const float*  xrow = x + (size_t)(row0 + l16) * CC;

  // Staging: 512 granules of 16B cover 128 rows x 32 elems; granule G:
  // h = G>>2, k-sub = (G&3)*8; LDS byte offset within buffer = G*16.
  int G0 = threadIdx.x * 2;
  int G1 = G0 + 1;
  const __bf16* g0 = Wm + (size_t)(G0 >> 2) * CC + (G0 & 3) * 8;
  const __bf16* g1 = Wm + (size_t)(G1 >> 2) * CC + (G1 & 3) * 8;
  unsigned ldsbase = (unsigned)(size_t)&wbuf[0][0];

  // Prologue: stage k0 = 0 into buffer 0.
  stage_granule16(g0, ldsbase + G0 * 16, &wbuf[0][G0 * 8]);
  stage_granule16(g1, ldsbase + G1 * 16, &wbuf[0][G1 * 8]);
  WAIT_ASYNC();
  __syncthreads();

  v8f acc[8];
#pragma unroll
  for (int j = 0; j < 8; ++j) acc[j] = (v8f){};

  for (int it = 0; it < CC / 32; ++it) {
    int k0 = it * 32;
    int buf = it & 1;
    if (it + 1 < CC / 32) {
      int nb = (it + 1) & 1;
      unsigned nbase = ldsbase + nb * (128 * 32 * 2);
      stage_granule16(g0 + k0 + 32, nbase + G0 * 16, &wbuf[nb][G0 * 8]);
      stage_granule16(g1 + k0 + 32, nbase + G1 * 16, &wbuf[nb][G1 * 8]);
    }

    // A fragment 16x32 bf16 (converted from f32 x): lane<16 holds K
    // {0..15}, lane>=16 holds K {16..31} of this 32-chunk.
    const float* pa = xrow + k0 + hi * 16;
    v4f a0 = *(const v4f*)(pa);
    v4f a1 = *(const v4f*)(pa + 4);
    v4f a2 = *(const v4f*)(pa + 8);
    v4f a3 = *(const v4f*)(pa + 12);
    v16bf a;
#pragma unroll
    for (int i = 0; i < 4; ++i) {
      a[i]      = (__bf16)a0[i];
      a[4 + i]  = (__bf16)a1[i];
      a[8 + i]  = (__bf16)a2[i];
      a[12 + i] = (__bf16)a3[i];
    }

    const __bf16* bp = &wbuf[buf][0];
#pragma unroll
    for (int j = 0; j < 8; ++j) {
      // B fragment 32x16: lane column h = j*16+l16, K split at 16 by lane half.
      v16bf b = *(const v16bf*)(bp + (j * 16 + l16) * 32 + hi * 16);
      acc[j] = __builtin_amdgcn_wmma_f32_16x16x32_bf16(
          false, a, false, b, (short)0, acc[j], false, false);
    }

    WAIT_ASYNC();
    __syncthreads();
  }

  const float* bias = (mat == 0) ? bq : ((mat == 1) ? bk : bv);

  if (mat < 2) {
    __bf16* dst = (mat == 0) ? Q : K;
#pragma unroll
    for (int j = 0; j < 8; ++j) {
      float bval = bias[j * 16 + l16];
#pragma unroll
      for (int r = 0; r < 8; ++r)
        dst[(size_t)(row0 + r + 8 * hi) * HH + j * 16 + l16] =
            (__bf16)(acc[j][r] + bval);
    }
  } else {
    int b    = row0 / TT;
    int tloc = row0 - b * TT;
#pragma unroll
    for (int j = 0; j < 8; ++j) {
      float bval = bias[j * 16 + l16];
#pragma unroll
      for (int r = 0; r < 8; ++r)
        Vt[(size_t)b * HH * TT + (size_t)(j * 16 + l16) * TT + tloc + r + 8 * hi] =
            (__bf16)(acc[j][r] + bval);
    }
  }
}

// ---------------------------------------------------------------------------
// Kernel 2: flash attention. One wave owns 16 query rows; online softmax over
// 32-key blocks (causal). P re-layout via wave-private LDS slice.
// ---------------------------------------------------------------------------
__global__ __launch_bounds__(128) void attn_kernel(
    const __bf16* __restrict__ Q, const __bf16* __restrict__ Km,
    const __bf16* __restrict__ Vt, float* __restrict__ out) {
  __shared__ __align__(16) __bf16 plds[4][16 * 32];   // 1KB per wave

  int w    = (threadIdx.x >> 5) & 3;
  int wid  = blockIdx.x * 4 + (threadIdx.x >> 5);     // 1024 wave-tiles
  int lane = threadIdx.x & 31;
  int l16  = lane & 15;
  int hi   = lane >> 4;

  int b  = wid >> 7;                                  // 128 tiles per batch
  int qt = wid & 127;
  int q0 = qt * 16;                                   // local query row base
  size_t rowg = (size_t)b * TT + q0;

  // Q fragments over H=128 (4 chunks of 32)
  v16bf qf[4];
  const __bf16* qrow = Q + (rowg + l16) * HH;
#pragma unroll
  for (int c = 0; c < 4; ++c) {
    v8bf lo = *(const v8bf*)(qrow + c * 32 + hi * 16);
    v8bf hh = *(const v8bf*)(qrow + c * 32 + hi * 16 + 8);
#pragma unroll
    for (int i = 0; i < 8; ++i) { qf[c][i] = lo[i]; qf[c][8 + i] = hh[i]; }
  }

  v8f o[8];
#pragma unroll
  for (int j = 0; j < 8; ++j) o[j] = (v8f){};
  float m[8], lsum[8];
#pragma unroll
  for (int r = 0; r < 8; ++r) { m[r] = -__builtin_inff(); lsum[r] = 0.0f; }

  const __bf16* Kbase = Km + (size_t)b * TT * HH;
  const __bf16* Vbase = Vt + (size_t)b * HH * TT;
  const float kscale = 0.08838834764831845f * 1.4426950408889634f; // H^-1/2 * log2(e)

  for (int kb = 0; kb <= q0 + 15; kb += 32) {
    v8f s[2];
#pragma unroll
    for (int t = 0; t < 2; ++t) {
      v8f sacc = (v8f){};
      const __bf16* krow = Kbase + (size_t)(kb + t * 16 + l16) * HH + hi * 16;
#pragma unroll
      for (int c = 0; c < 4; ++c) {
        v16bf bf = *(const v16bf*)(krow + c * 32);
        sacc = __builtin_amdgcn_wmma_f32_16x16x32_bf16(
            false, qf[c], false, bf, (short)0, sacc, false, false);
      }
      int col = kb + t * 16 + l16;
#pragma unroll
      for (int r = 0; r < 8; ++r) {
        int rowr = q0 + r + 8 * hi;
        float v = sacc[r] * kscale;
        s[t][r] = (col > rowr) ? -__builtin_inff() : v;
      }
    }

    // online softmax: row max (16-lane halves match C-fragment rows)
    float alpha[8];
#pragma unroll
    for (int r = 0; r < 8; ++r) {
      float t0 = fmaxf(s[0][r], s[1][r]);
#pragma unroll
      for (int d = 1; d < 16; d <<= 1)
        t0 = fmaxf(t0, __shfl_xor(t0, d, 32));
      float mnew = fmaxf(m[r], t0);
      alpha[r] = EXP2F(m[r] - mnew);    // exp2(-inf) = 0 on first block
      m[r] = mnew;
    }
#pragma unroll
    for (int t = 0; t < 2; ++t)
#pragma unroll
      for (int r = 0; r < 8; ++r) s[t][r] = EXP2F(s[t][r] - m[r]);

#pragma unroll
    for (int r = 0; r < 8; ++r) {
      float t0 = s[0][r] + s[1][r];
#pragma unroll
      for (int d = 1; d < 16; d <<= 1)
        t0 += __shfl_xor(t0, d, 32);
      lsum[r] = lsum[r] * alpha[r] + t0;
    }
#pragma unroll
    for (int j = 0; j < 8; ++j)
#pragma unroll
      for (int r = 0; r < 8; ++r) o[j][r] *= alpha[r];

    // P (C-layout f32) -> bf16 A-layout via wave-private LDS
    __bf16* lp = plds[w];
#pragma unroll
    for (int t = 0; t < 2; ++t)
#pragma unroll
      for (int r = 0; r < 8; ++r)
        lp[(r + 8 * hi) * 32 + t * 16 + l16] = (__bf16)s[t][r];

    v16bf pf;
    {
      v8bf lo = *(const v8bf*)(lp + l16 * 32 + hi * 16);
      v8bf hh = *(const v8bf*)(lp + l16 * 32 + hi * 16 + 8);
#pragma unroll
      for (int i = 0; i < 8; ++i) { pf[i] = lo[i]; pf[8 + i] = hh[i]; }
    }

    // O += P @ V  (V transposed layout -> contiguous B-fragment loads)
#pragma unroll
    for (int j = 0; j < 8; ++j) {
      const __bf16* vrow = Vbase + (size_t)(j * 16 + l16) * TT + kb + hi * 16;
      v16bf vf = *(const v16bf*)(vrow);
      o[j] = __builtin_amdgcn_wmma_f32_16x16x32_bf16(
          false, pf, false, vf, (short)0, o[j], false, false);
    }
  }

  float rl[8];
#pragma unroll
  for (int r = 0; r < 8; ++r) rl[r] = 1.0f / lsum[r];
#pragma unroll
  for (int j = 0; j < 8; ++j)
#pragma unroll
    for (int r = 0; r < 8; ++r)
      out[(rowg + r + 8 * hi) * HH + j * 16 + l16] = o[j][r] * rl[r];
}

// ---------------------------------------------------------------------------
extern "C" void kernel_launch(void* const* d_in, const int* in_sizes, int n_in,
                              void* d_out, int out_size, void* d_ws, size_t ws_size,
                              hipStream_t stream) {
  const float* x  = (const float*)d_in[0];
  const float* Wq = (const float*)d_in[1];
  const float* bq = (const float*)d_in[2];
  const float* Wk = (const float*)d_in[3];
  const float* bk = (const float*)d_in[4];
  const float* Wv = (const float*)d_in[5];
  const float* bv = (const float*)d_in[6];
  float* out = (float*)d_out;

  __bf16* ws = (__bf16*)d_ws;
  const size_t qkvN = (size_t)BB * TT * HH;   // 2,097,152 elems each
  __bf16* Q  = ws;
  __bf16* K  = Q + qkvN;
  __bf16* Vt = K + qkvN;
  __bf16* Wt = Vt + qkvN;                     // 3*C*H elems

  // K0: transpose weights (3*C*H = 786432 elements)
  wtrans_kernel<<<(3 * CC * HH) / 256, 256, 0, stream>>>(Wq, Wk, Wv, Wt);
  // K1: QKV projection (384 blocks = 3 mats x 128 row-groups, 8 waves each)
  qkv_kernel<<<384, 256, 0, stream>>>(x, Wt, bq, bk, bv, Q, K, Vt);
  // K2: flash attention (1024 wave-tiles / 4 waves per block)
  attn_kernel<<<256, 128, 0, stream>>>(Q, K, Vt, out);
}
